// SupConLoss_k_50379966382589
// MI455X (gfx1250) — matrix operand order repaired and checked by reading, not compiled, not run on hardware
//
#include <hip/hip_runtime.h>
#include <hip/hip_bf16.h>

// ---------------------------------------------------------------------------
// SupConLoss fused kernel for MI455X (gfx1250, wave32, WMMA, TDM)
//
//  1) convert_kernel : features fp32 [B,V,D] -> cf bf16 [V*B, D] (4 MB, L2-resident)
//  2) select_kernel  : per-anchor random positive via hash-argmax
//  3) gemm_stats     : fused Gram-matrix (bf16 WMMA, fp32 acc, 2 C-tiles/wave)
//                      + lane-local online softmax stats (branch-free mask),
//                      one cross-lane combine at end
//  4) finalize       : combine 16 column-slice partials, mean -> d_out[0]
// ---------------------------------------------------------------------------

#define B_N   2048
#define V_N   2
#define VB_N  4096
#define D_N   512
#define INV_T (1.0f / 0.07f)

typedef __attribute__((ext_vector_type(16))) __bf16        v16bf;
typedef __attribute__((ext_vector_type(8)))  float         v8f;
typedef __attribute__((ext_vector_type(4)))  unsigned int  uiv4;
typedef __attribute__((ext_vector_type(8)))  int           iv8;
typedef __attribute__((ext_vector_type(4)))  int           iv4;

union ABU { v16bf v; uiv4 q[2]; };  // 32-byte A/B fragment (16 bf16)

static __device__ __forceinline__ unsigned short f2bf(float f) {
    unsigned u = __float_as_uint(f);
    if ((u & 0x7fffffffu) > 0x7f800000u) return (unsigned short)((u >> 16) | 0x0040u); // NaN
    unsigned r = u + 0x7fffu + ((u >> 16) & 1u);  // round-to-nearest-even
    return (unsigned short)(r >> 16);
}

static __device__ __forceinline__ unsigned hashmix(unsigned x) {
    x ^= x >> 16; x *= 0x7feb352du;
    x ^= x >> 15; x *= 0x846ca68bu;
    x ^= x >> 16; return x;
}

// --------------------------- 1) fp32 -> bf16 cf ----------------------------
__global__ __launch_bounds__(256) void convert_kernel(const float* __restrict__ f,
                                                      unsigned short* __restrict__ cf) {
    int idx = blockIdx.x * 256 + threadIdx.x;         // idx = (b*V + v)*D + d
    int d  = idx & (D_N - 1);
    int bv = idx >> 9;
    int v  = bv & 1;
    int b  = bv >> 1;
    cf[((size_t)(v * B_N + b)) * D_N + d] = f2bf(f[idx]);
}

// --------------------- 2) random positive selection ------------------------
__global__ __launch_bounds__(256) void select_kernel(const int* __restrict__ mask_lab,
                                                     const int* __restrict__ labels,
                                                     int* __restrict__ chosen,
                                                     int* __restrict__ has) {
    const int b   = blockIdx.x;
    const int tid = threadIdx.x;
    __shared__ unsigned sS[256];
    __shared__ int      sI[256];
    __shared__ int      sC[256];
    const int lb = labels[b];
    const int ml = mask_lab[b];
    unsigned best = 0; int bi = -1; int cnt = 0;
    for (int j = tid; j < B_N; j += 256) {
        int cand = (int)(labels[j] == lb) & (int)(j != b) & ((int)!ml | mask_lab[j]);
        if (cand) {
            unsigned h = hashmix(((unsigned)b * 2048u + (unsigned)j) ^ 0x2545F491u) | 1u;
            ++cnt;
            if (h > best) { best = h; bi = j; }
        }
    }
    sS[tid] = best; sI[tid] = bi; sC[tid] = cnt;
    __syncthreads();
    for (int s = 128; s > 0; s >>= 1) {
        if (tid < s) {
            if (sS[tid + s] > sS[tid]) { sS[tid] = sS[tid + s]; sI[tid] = sI[tid + s]; }
            sC[tid] += sC[tid + s];
        }
        __syncthreads();
    }
    if (tid == 0) { chosen[b] = sI[0]; has[b] = (sC[0] > 0) ? 1 : 0; }
}

// ----------- 3) fused WMMA Gram product + online softmax stats -------------
__global__ __launch_bounds__(256) void gemm_stats_kernel(const unsigned short* __restrict__ cf,
                                                         const int* __restrict__ chosen,
                                                         const int* __restrict__ has,
                                                         float* __restrict__ part) {
    __shared__ unsigned short ldsA[64 * D_N];   // 64 KB: A row-block, all K

    const int tid  = threadIdx.x;
    const int wave = tid >> 5, lane = tid & 31;
    const int rt   = wave >> 1, ct = wave & 1;          // 4 row tiles x 2 col streams
    const int half = lane >> 4, ln = lane & 15;
    const int rowBase   = blockIdx.x * 64;
    const int sliceBase = blockIdx.y * 512;

    // ---- stage A (64 rows x 512 K, bf16) into LDS via the Tensor Data Mover ----
#if defined(__gfx1250__) && __has_builtin(__builtin_amdgcn_tensor_load_to_lds)
    if (wave == 0) {
        // TDM 2D descriptor: dim0 = K (contig, 512 elems of 2B), dim1 = 64 rows.
        unsigned long long ga =
            (unsigned long long)(uintptr_t)(cf + (size_t)rowBase * D_N);
        uiv4 g0 = { 1u,                                  // count=1 (valid descriptor)
                    0u,                                  // lds_addr = 0 (ldsA is sole LDS obj)
                    (unsigned)(ga & 0xffffffffu),
                    (unsigned)((ga >> 32) & 0x01ffffffu) | (2u << 30) }; // addr hi | type=2
        iv8  g1 = { (int)0x00010000u,                    // data_size=1 (2B), no multicast
                    (int)(512u << 16),                   // tensor_dim0 = 512
                    (int)(4096u << 16),                  // tensor_dim1 = 4096
                    (int)(512u << 16),                   // tile_dim0 = 512
                    64,                                  // tile_dim1 = 64
                    512,                                 // tensor_dim0_stride = 512
                    0, 0 };
        iv4  gz4 = { 0, 0, 0, 0 };
        iv8  gz8 = { 0, 0, 0, 0, 0, 0, 0, 0 };
        __builtin_amdgcn_tensor_load_to_lds(g0, g1, gz4, gz4, gz8, 0);
        __builtin_amdgcn_s_wait_tensorcnt(0);
    }
#else
    {
        const uiv4* src = (const uiv4*)(cf + (size_t)rowBase * D_N);
        uiv4*       dst = (uiv4*)ldsA;
        for (int i = tid; i < 64 * D_N / 8; i += 256) dst[i] = src[i];
    }
#endif
    // TDM writes LDS through a descriptor the compiler can't see; escape the
    // LDS object so the A-fragment ds_loads are not folded to undef.
    asm volatile("" : : "r"(&ldsA[0]) : "memory");
    __syncthreads();

    // ---- per-lane row state: C-fragment VGPR r of this lane is row r + 8*half ----
    float m[8], S[8], P[8], C[8];
    int   ch[8], hs[8], rowAbs[8], b1r[8];
#pragma unroll
    for (int r = 0; r < 8; ++r) {
        int i = rowBase + rt * 16 + r + 8 * half;
        rowAbs[r] = i;
        b1r[r] = i & (B_N - 1);
        ch[r] = chosen[b1r[r]]; hs[r] = has[b1r[r]];
        m[r] = -INFINITY; S[r] = 0.f; P[r] = 0.f; C[r] = 0.f;
    }

    // Branch-free per-element online update: one v_exp, cndmask selects, and a
    // bitwise big_mask predicate (no exec-mask control flow).
    auto upd = [&](int r, float x, int j, int b2) {
        float mold = m[r];
        float mnew = fmaxf(mold, x);
        float ed   = __expf(fminf(mold, x) - mnew);   // exp(-|x-mold|); 0 on first col
        int   diag = (int)(rowAbs[r] == j);
        int   xs   = (int)(x <= mold);
        float mul  = xs ? 1.0f : ed;
        float addv = (diag | 0) ? 0.0f : (xs ? ed : 1.0f);
        S[r] = fmaf(S[r], mul, addv);
        m[r] = mnew;
        int bmI = ((int)(b2 == b1r[r]) | (hs[r] & (int)(ch[r] == b2))) & (diag ^ 1);
        float bm = (float)bmI;
        P[r] = fmaf(bm, x, P[r]);
        C[r] += bm;
    };

    const int aRowOff = (rt * 16 + ln) * D_N;  // A: lane -> matrix row, halves split K
    for (int iter = 0; iter < 8; ++iter) {
        const int colBase = sliceBase + iter * 64 + ct * 32;   // wave owns 32 cols
        const int bRow0   = colBase + ln;                       // B: lane -> column
        const int bRow1   = colBase + 16 + ln;
        v8f acc0 = {}, acc1 = {};
#pragma unroll 4
        for (int ks = 0; ks < 16; ++ks) {
            const int kk = ks * 32;
            ABU a, b0, b1;
            // A 16x32 bf16 layout: lane half h holds K {kk+16h .. +15} contiguously
            a.q[0] = *(const uiv4*)&ldsA[aRowOff + kk + 16 * half];
            a.q[1] = *(const uiv4*)&ldsA[aRowOff + kk + 16 * half + 8];
            // B 32x16 bf16 layout: lane half h holds K {kk+16h .. +15} contiguously
            const unsigned short* bp0 = cf + (size_t)bRow0 * D_N + kk + 16 * half;
            const unsigned short* bp1 = cf + (size_t)bRow1 * D_N + kk + 16 * half;
            b0.q[0] = *(const uiv4*)(bp0); b0.q[1] = *(const uiv4*)(bp0 + 8);
            b1.q[0] = *(const uiv4*)(bp1); b1.q[1] = *(const uiv4*)(bp1 + 8);
            acc0 = __builtin_amdgcn_wmma_f32_16x16x32_bf16(
                false, a.v, false, b0.v, (short)0, acc0, false, false);
            acc1 = __builtin_amdgcn_wmma_f32_16x16x32_bf16(
                false, a.v, false, b1.v, (short)0, acc1, false, false);
        }

        const int j0 = colBase + ln;
        const int j1 = j0 + 16;
        const int c0 = j0 & (B_N - 1);
        const int c1 = j1 & (B_N - 1);
#pragma unroll
        for (int r = 0; r < 8; ++r) {
            upd(r, acc0[r] * INV_T, j0, c0);
            upd(r, acc1[r] * INV_T, j1, c1);
        }
    }

    // ---- single cross-lane combine (width-16 butterflies, log-sum-exp merge) ----
#pragma unroll
    for (int r = 0; r < 8; ++r) {
#pragma unroll
        for (int off = 1; off <= 8; off <<= 1) {
            float mo = __shfl_xor(m[r], off, 16);
            float So = __shfl_xor(S[r], off, 16);
            float mn = fmaxf(m[r], mo);
            float ed = __expf(fminf(m[r], mo) - mn);
            S[r] = (m[r] <= mo) ? fmaf(S[r], ed, So) : fmaf(So, ed, S[r]);
            m[r] = mn;
            P[r] += __shfl_xor(P[r], off, 16);
            C[r] += __shfl_xor(C[r], off, 16);
        }
    }

    if (ln == 0) {  // lanes 0 and 16 each own 8 distinct rows
        const int slot = blockIdx.y * 2 + ct;   // 16 partial slots total
#pragma unroll
        for (int r = 0; r < 8; ++r) {
            float* q = part + ((size_t)slot * VB_N + rowAbs[r]) * 4;
            q[0] = m[r]; q[1] = S[r]; q[2] = P[r]; q[3] = C[r];
        }
    }
}

// ----------------------- 4) combine partials, mean -------------------------
__global__ __launch_bounds__(256) void finalize_kernel(const float* __restrict__ part,
                                                       float* __restrict__ out) {
    __shared__ float red[256];
    const int tid = threadIdx.x;
    float local = 0.f;
    for (int k = 0; k < VB_N / 256; ++k) {
        const int i = tid + 256 * k;
        float mm = -INFINITY;
        for (int s = 0; s < 16; ++s)
            mm = fmaxf(mm, part[((size_t)s * VB_N + i) * 4]);
        float S = 0.f, P = 0.f, Cc = 0.f;
        for (int s = 0; s < 16; ++s) {
            const float* q = part + ((size_t)s * VB_N + i) * 4;
            S  += q[1] * __expf(q[0] - mm);
            P  += q[2];
            Cc += q[3];
        }
        // loss_i = -(T/baseT) * (P/C - m - log S), T/baseT == 1
        local += -(P / Cc - mm - __logf(S));
    }
    red[tid] = local;
    __syncthreads();
    for (int s = 128; s > 0; s >>= 1) {
        if (tid < s) red[tid] += red[tid + s];
        __syncthreads();
    }
    if (tid == 0) out[0] = red[0] / (float)VB_N;
}

// ---------------------------------------------------------------------------
extern "C" void kernel_launch(void* const* d_in, const int* in_sizes, int n_in,
                              void* d_out, int out_size, void* d_ws, size_t ws_size,
                              hipStream_t stream) {
    const float* features = (const float*)d_in[0];
    const int*   mask_lab = (const int*)d_in[1];
    const int*   labels   = (const int*)d_in[2];
    float*       out      = (float*)d_out;

    char* ws = (char*)d_ws;
    // workspace layout (~5.3 MB total)
    unsigned short* cf     = (unsigned short*)ws;                       // 4 MB
    int*            chosen = (int*)(ws + (size_t)VB_N * D_N * 2);       // 8 KB
    int*            has    = chosen + B_N;                              // 8 KB
    float*          part   = (float*)(ws + (size_t)VB_N * D_N * 2 + 16384); // 1 MB

    convert_kernel<<<(B_N * V_N * D_N) / 256, 256, 0, stream>>>(features, cf);
    select_kernel<<<B_N, 256, 0, stream>>>(mask_lab, labels, chosen, has);
    gemm_stats_kernel<<<dim3(64, 8, 1), 256, 0, stream>>>(cf, chosen, has, part);
    finalize_kernel<<<1, 256, 0, stream>>>(part, out);
}
